// self_attention_58643483459816
// MI455X (gfx1250) — compile-verified, matrix operand order checked by reading, not run on hardware
//
#include <hip/hip_runtime.h>
#include <hip/hip_bf16.h>

typedef __attribute__((ext_vector_type(16))) _Float16 v16h;
typedef __attribute__((ext_vector_type(8)))  _Float16 v8h;
typedef __attribute__((ext_vector_type(8)))  float    v8f;
typedef __attribute__((ext_vector_type(4)))  float    v4f;

#define BATCH 4
#define CHAN  64
#define CQK   8
#define NPIX  4096            // 64*64
#define SSTR  4100            // padded LDS row stride (floats) to break bank conflicts
#define QROW  16              // q/kt rows zero-padded to 16 halves (32B) in memory

// ---------------------------------------------------------------------------
// Kernel 1: per-pixel 1x1 conv projections.
//   q  -> (B, N, 16) f16, halves 0..7 = data, 8..15 = 0  (branch-free A operand)
//   kt -> (B, N, 16) f16, same padding                    (branch-free B operand)
//   v  -> (B, 64, N) f16 (m-contiguous: B-operand friendly for P*V^T)
// ---------------------------------------------------------------------------
__global__ __launch_bounds__(256) void sa_proj_kernel(
    const float* __restrict__ x, const float* __restrict__ y,
    const float* __restrict__ Wq, const float* __restrict__ bq,
    const float* __restrict__ Wk, const float* __restrict__ bk,
    const float* __restrict__ Wv, const float* __restrict__ bv,
    _Float16* __restrict__ qbuf, _Float16* __restrict__ ktbuf,
    _Float16* __restrict__ vbuf)
{
  __shared__ float sWq[CQK * CHAN];
  __shared__ float sWk[CQK * CHAN];
  __shared__ float sWv[CHAN * CHAN];
  const int tid = threadIdx.x;
  for (int i = tid; i < CQK * CHAN; i += 256) { sWq[i] = Wq[i]; sWk[i] = Wk[i]; }
  for (int i = tid; i < CHAN * CHAN; i += 256) sWv[i] = Wv[i];
  __syncthreads();

  const int g = blockIdx.x * 256 + tid;     // global pixel id over B*N
  const int b = g >> 12;
  const int n = g & (NPIX - 1);

  v8h zero8;
#pragma unroll
  for (int i = 0; i < 8; ++i) zero8[i] = (_Float16)0.f;

  float xv[CHAN];
  {
    const float* xp = x + ((size_t)b * CHAN) * NPIX + n;
#pragma unroll
    for (int c = 0; c < CHAN; ++c) xv[c] = xp[(size_t)c * NPIX];
  }

  // q = Wq x + bq   (8 outputs, zero-padded row)
  {
    v8h qh;
#pragma unroll
    for (int o = 0; o < CQK; ++o) {
      float acc = bq[o];
#pragma unroll
      for (int c = 0; c < CHAN; ++c) acc += sWq[o * CHAN + c] * xv[c];
      qh[o] = (_Float16)acc;
    }
    *(v8h*)(qbuf + (size_t)g * QROW)     = qh;
    *(v8h*)(qbuf + (size_t)g * QROW + 8) = zero8;
  }

  // v = Wv x + bv   (64 outputs, stored channel-major / m-contiguous)
  for (int c = 0; c < CHAN; ++c) {
    float acc = bv[c];
#pragma unroll
    for (int cc = 0; cc < CHAN; ++cc) acc += sWv[c * CHAN + cc] * xv[cc];
    vbuf[((size_t)b * CHAN + c) * NPIX + n] = (_Float16)acc;
  }

  // k = Wk y + bk   (8 outputs, stored transposed as (N,16) zero-padded)
  {
    const float* yp = y + ((size_t)b * CHAN) * NPIX + n;
#pragma unroll
    for (int c = 0; c < CHAN; ++c) xv[c] = yp[(size_t)c * NPIX];
    v8h kh;
#pragma unroll
    for (int o = 0; o < CQK; ++o) {
      float acc = bk[o];
#pragma unroll
      for (int c = 0; c < CHAN; ++c) acc += sWk[o * CHAN + c] * xv[c];
      kh[o] = (_Float16)acc;
    }
    *(v8h*)(ktbuf + (size_t)g * QROW)     = kh;
    *(v8h*)(ktbuf + (size_t)g * QROW + 8) = zero8;
  }
}

// ---------------------------------------------------------------------------
// Kernel 2: xsum[b][c] = sum_n x[b,c,n]   (for the Sv closed form)
// ---------------------------------------------------------------------------
__global__ __launch_bounds__(256) void sa_xsum_kernel(
    const float* __restrict__ x, float* __restrict__ xsum)
{
  __shared__ float red[256];
  const int bc = blockIdx.x;               // b*64 + c
  const float* p = x + (size_t)bc * NPIX;
  float s = 0.f;
  for (int i = threadIdx.x; i < NPIX; i += 256) s += p[i];
  red[threadIdx.x] = s;
  __syncthreads();
  for (int off = 128; off; off >>= 1) {
    if (threadIdx.x < off) red[threadIdx.x] += red[threadIdx.x + off];
    __syncthreads();
  }
  if (threadIdx.x == 0) xsum[bc] = red[0];
}

// ---------------------------------------------------------------------------
// Kernel 3: Sv[b][c] = Wv[c,:] . xsum[b,:] + N * bv[c]
// ---------------------------------------------------------------------------
__global__ __launch_bounds__(256) void sa_sv_kernel(
    const float* __restrict__ Wv, const float* __restrict__ bv,
    const float* __restrict__ xsum, float* __restrict__ sv)
{
  const int tid = threadIdx.x;             // 256 threads = B*C
  const int b = tid >> 6, c = tid & 63;
  float acc = (float)NPIX * bv[c];
#pragma unroll
  for (int cc = 0; cc < CHAN; ++cc) acc += Wv[c * CHAN + cc] * xsum[b * CHAN + cc];
  sv[tid] = acc;
}

// ---------------------------------------------------------------------------
// Kernel 4: fused attention. One workgroup (8 wave32) per (batch, 16-row tile).
//  Phase A: S = Q K^T via v_wmma_f32_16x16x32_f16 (K padded 8->32) -> LDS,
//           with per-row running max fused in (registers + shuffle reduce)
//  Phase B: single exp+rowsum sweep (max already known), P kept unnormalized
//  Phase C: O = P V^T via WMMA, K-split across 8 waves, LDS reduce
//  Epilogue: out = gamma*(Sv - O/rowsum) + x
// ---------------------------------------------------------------------------
__global__ __launch_bounds__(256) void sa_attn_kernel(
    const _Float16* __restrict__ qbuf, const _Float16* __restrict__ ktbuf,
    const _Float16* __restrict__ vbuf, const float* __restrict__ sv,
    const float* __restrict__ x, const float* __restrict__ gamma,
    float* __restrict__ out)
{
  extern __shared__ float smem[];
  float* S       = smem;                    // 16 * SSTR floats
  float* rowsum  = S + 16 * SSTR;           // 16 floats
  float* rowmaxp = rowsum + 16;             // 8 * 16 floats (per-wave row-max)
  float* Opart   = rowmaxp + 128;           // 8 * 16 * 64 floats

  const int tid  = threadIdx.x;
  const int lane = tid & 31;
  const int w    = tid >> 5;
  const int b    = blockIdx.x >> 8;        // 256 tiles per batch
  const int j0   = (blockIdx.x & 255) * 16;

  const int halfrow = (lane >= 16) ? 8 : 0;   // C/D layout: upper half-wave -> rows 8..15
  const int l15     = lane & 15;
  const int kbo     = (lane >> 4) * 8;        // A/B f16 layout K base per half-wave

  // ----- A operand: Q tile (16x8 valid, padded to 16x32); branch-free -----
  v16h qa;
#pragma unroll
  for (int i = 8; i < 16; ++i) qa[i] = (_Float16)0.f;   // K>=16 always zero
  {
    v8h t = *(const v8h*)(qbuf + ((size_t)b * NPIX + j0 + l15) * QROW + kbo);
#pragma unroll
    for (int i = 0; i < 8; ++i) qa[i] = t[i];
  }

  // ----- Phase A: scores into LDS + fused running row max -----
  float vmax[8];
#pragma unroll
  for (int r = 0; r < 8; ++r) vmax[r] = -1e30f;

  for (int t = w * 32; t < w * 32 + 32; ++t) {
    const int m0 = t * 16;
    v16h kb;
#pragma unroll
    for (int i = 8; i < 16; ++i) kb[i] = (_Float16)0.f;
    v8h kk = *(const v8h*)(ktbuf + ((size_t)b * NPIX + m0 + l15) * QROW + kbo);
#pragma unroll
    for (int i = 0; i < 8; ++i) kb[i] = kk[i];

    v8f c;
#pragma unroll
    for (int i = 0; i < 8; ++i) c[i] = 0.f;
    c = __builtin_amdgcn_wmma_f32_16x16x32_f16(false, qa, false, kb,
                                               (short)0, c, false, false);
#pragma unroll
    for (int r = 0; r < 8; ++r) {
      S[(halfrow + r) * SSTR + m0 + l15] = c[r];
      vmax[r] = fmaxf(vmax[r], c[r]);
    }
  }
  // cross-lane max within each 16-lane half (rows halfrow..halfrow+7)
#pragma unroll
  for (int r = 0; r < 8; ++r) {
#pragma unroll
    for (int off = 1; off < 16; off <<= 1) vmax[r] = fmaxf(vmax[r], __shfl_xor(vmax[r], off));
  }
  if (l15 == 0) {
#pragma unroll
    for (int r = 0; r < 8; ++r) rowmaxp[w * 16 + halfrow + r] = vmax[r];
  }
  __syncthreads();

  // ----- Phase B: single exp + rowsum sweep (2 rows per wave) -----
  for (int j = w * 2; j < w * 2 + 2; ++j) {
    float m = -1e30f;
#pragma unroll
    for (int ww = 0; ww < 8; ++ww) m = fmaxf(m, rowmaxp[ww * 16 + j]);
    float s = 0.f;
    for (int i = lane; i < NPIX; i += 32) {
      float e = __expf(S[j * SSTR + i] - m);
      S[j * SSTR + i] = e;                 // unnormalized P
      s += e;
    }
#pragma unroll
    for (int off = 16; off; off >>= 1) s += __shfl_xor(s, off);
    if (lane == 0) rowsum[j] = s;
  }
  __syncthreads();

  // ----- Phase C: O += P * V^T, each wave owns 512 keys -----
  v8f acc[4];
#pragma unroll
  for (int cc = 0; cc < 4; ++cc)
#pragma unroll
    for (int i = 0; i < 8; ++i) acc[cc][i] = 0.f;

  const _Float16* vbase = vbuf + (size_t)b * CHAN * NPIX;
  for (int chunk = 0; chunk < 16; ++chunk) {
    const int m0 = w * 512 + chunk * 32;
    // A operand from LDS (f32 -> f16 on the fly)
    const float* pr = S + l15 * SSTR + m0 + kbo;
    v4f f0 = *(const v4f*)pr;
    v4f f1 = *(const v4f*)(pr + 4);
    v4f f2 = *(const v4f*)(pr + 16);
    v4f f3 = *(const v4f*)(pr + 20);
    v16h pa;
#pragma unroll
    for (int i = 0; i < 4; ++i) {
      pa[i]      = (_Float16)f0[i];
      pa[4 + i]  = (_Float16)f1[i];
      pa[8 + i]  = (_Float16)f2[i];
      pa[12 + i] = (_Float16)f3[i];
    }
#pragma unroll
    for (int cc = 0; cc < 4; ++cc) {
      const int c = cc * 16 + l15;
      const _Float16* vp = vbase + (size_t)c * NPIX + m0 + kbo;
      v8h v0 = *(const v8h*)vp;
      v8h v1 = *(const v8h*)(vp + 16);
      v16h vb;
#pragma unroll
      for (int i = 0; i < 8; ++i) { vb[i] = v0[i]; vb[8 + i] = v1[i]; }
      acc[cc] = __builtin_amdgcn_wmma_f32_16x16x32_f16(false, pa, false, vb,
                                                       (short)0, acc[cc],
                                                       false, false);
    }
  }
  // stash per-wave partials
#pragma unroll
  for (int cc = 0; cc < 4; ++cc)
#pragma unroll
    for (int r = 0; r < 8; ++r)
      Opart[w * 1024 + (halfrow + r) * 64 + cc * 16 + l15] = acc[cc][r];
  __syncthreads();

  // ----- Epilogue: out = gamma*(Sv - O/rowsum) + x -----
  const float g = gamma[0];
  for (int e = tid; e < 16 * 64; e += 256) {
    const int j = e >> 6, c = e & 63;
    float o = 0.f;
#pragma unroll
    for (int ww = 0; ww < 8; ++ww) o += Opart[ww * 1024 + e];
    const float val = sv[b * CHAN + c] - o / rowsum[j];
    const size_t oi = ((size_t)b * CHAN + c) * NPIX + j0 + j;
    out[oi] = g * val + x[oi];
  }
}

// ---------------------------------------------------------------------------
extern "C" void kernel_launch(void* const* d_in, const int* in_sizes, int n_in,
                              void* d_out, int out_size, void* d_ws, size_t ws_size,
                              hipStream_t stream) {
  const float* x     = (const float*)d_in[0];
  const float* y     = (const float*)d_in[1];
  const float* Wq    = (const float*)d_in[2];
  const float* bq    = (const float*)d_in[3];
  const float* Wk    = (const float*)d_in[4];
  const float* bk    = (const float*)d_in[5];
  const float* Wv    = (const float*)d_in[6];
  const float* bv    = (const float*)d_in[7];
  const float* gamma = (const float*)d_in[8];
  float* out = (float*)d_out;

  // workspace carve-up
  char* ws = (char*)d_ws;
  _Float16* qbuf  = (_Float16*)ws;                               // B*N*16 f16 = 512 KB
  _Float16* ktbuf = (_Float16*)(ws + 524288);                    // B*N*16 f16 = 512 KB
  _Float16* vbuf  = (_Float16*)(ws + 1048576);                   // B*64*N f16 = 2 MB
  float*    xsum  = (float*)   (ws + 3145728);                   // B*64 f32
  float*    svb   = (float*)   (ws + 3146752);                   // B*64 f32

  const int smem_bytes =
      (16 * SSTR + 16 + 128 + 8 * 16 * 64) * sizeof(float);      // ~289 KB
  hipFuncSetAttribute((const void*)sa_attn_kernel,
                      hipFuncAttributeMaxDynamicSharedMemorySize, smem_bytes);

  sa_proj_kernel<<<(BATCH * NPIX) / 256, 256, 0, stream>>>(
      x, y, Wq, bq, Wk, bk, Wv, bv, qbuf, ktbuf, vbuf);
  sa_xsum_kernel<<<BATCH * CHAN, 256, 0, stream>>>(x, xsum);
  sa_sv_kernel<<<1, 256, 0, stream>>>(Wv, bv, xsum, svb);
  sa_attn_kernel<<<BATCH * (NPIX / 16), 256, smem_bytes, stream>>>(
      qbuf, ktbuf, vbuf, svb, x, gamma, out);
}
